// Loss_60576218742891
// MI455X (gfx1250) — matrix operand order repaired and checked by reading, not compiled, not run on hardware
//
#include <hip/hip_runtime.h>

// ---------------------------------------------------------------------------
// Loss kernel suite for MI455X (gfx1250, wave32, WMMA).
//
// Pipeline (all on `stream`, one f64 accumulator in d_ws):
//   K0  init accumulator
//   K1  guided-filter horizontal box of (x, x^2)         raw -> B0,B1
//   K2  vertical box + A/b math                          B0,B1 -> B2,B3
//   K3  horizontal box of (A, b)                         B2,B3 -> B0,B1
//   K4  vertical box + q + sum|smoothedTarget - q|       -> acc (w=1)
//   K5  sampled L1 * map (::8,::8)                       -> acc (w=1)
//   K6  TV terms for GNN and LR                          -> acc (w=0.1)
//   K7  Hessian 13x13 convs via v_wmma_f32_16x16x32_bf16 -> acc (w=0.5, Gxy x2)
//   K8  d_out[0] = (float)acc
// ---------------------------------------------------------------------------

#define HW   3072
#define NPIX (HW * HW)
#define R    29
#define WIN  (2 * R + 1)            // 59
#define KS   13
#define TAPS (KS * KS)              // 169
#define OW   (HW - KS + 1)          // 3060 valid conv outputs per dim

// Hessian WMMA tiling: K padded to 13 rows x 16 cols per source (208),
// two sources (x, x-t) -> K = 416 = 13 blocks of 32.
#define KROWPAD   16
#define KPERSRC   (KS * KROWPAD)    // 208
#define KBLK      13                // 416 / 32
#define TILE_COLS 31                // 16 outputs + 15 halo (padded cols)
#define TILE_ROWS 20                // 8 output rows + 12 halo
#define TILE_ELEMS (TILE_ROWS * TILE_COLS)
#define NTJ       8                 // column-tiles swept per block

typedef __attribute__((ext_vector_type(16))) __bf16 v16bf;
typedef __attribute__((ext_vector_type(8)))  float  v8f;

__device__ __forceinline__ unsigned short f32_to_bf16_bits(float f) {
  union { float f; unsigned u; } v; v.f = f;
  unsigned u = v.u;
  u += 0x7FFFu + ((u >> 16) & 1u);   // round-to-nearest-even
  return (unsigned short)(u >> 16);
}
__device__ __forceinline__ float bf16_bits_to_f32(unsigned short b) {
  union { unsigned u; float f; } v; v.u = ((unsigned)b) << 16;
  return v.f;
}
__device__ __forceinline__ __bf16 bits_to_bf16(unsigned short b) {
  return __builtin_bit_cast(__bf16, b);
}

// Block-wide (256 threads) reduction into the global f64 accumulator.
__device__ __forceinline__ void reduce_add_256(float local, double w,
                                               double* acc, float* red) {
  const int tid = threadIdx.x;
  red[tid] = local;
  __syncthreads();
  #pragma unroll
  for (int s = 128; s > 0; s >>= 1) {
    if (tid < s) red[tid] += red[tid + s];
    __syncthreads();
  }
  if (tid == 0) atomicAdd(acc, w * (double)red[0]);
}

// ------------------------------- K0 / K8 -----------------------------------
__global__ void k_init_acc(double* acc) {
  if (threadIdx.x == 0 && blockIdx.x == 0) *acc = 0.0;
}
__global__ void k_finalize(const double* acc, float* out) {
  if (threadIdx.x == 0 && blockIdx.x == 0) out[0] = (float)(*acc);
}

// ------------------------- guided filter passes ----------------------------
// Horizontal 59-wide box sum of x and x^2 (zero pad), one row chunk per block.
__global__ __launch_bounds__(256) void k_gf_hpass_sq(
    const float* __restrict__ x, float* __restrict__ oS, float* __restrict__ oS2) {
  __shared__ float sx[256 + WIN - 1];
  const int row = blockIdx.y, j0 = blockIdx.x * 256, tid = threadIdx.x;
  for (int e = tid; e < 256 + WIN - 1; e += 256) {
    int gj = j0 - R + e;
    sx[e] = (gj >= 0 && gj < HW) ? x[row * HW + gj] : 0.0f;
  }
  __syncthreads();
  float s1 = 0.0f, s2 = 0.0f;
  for (int d = 0; d < WIN; ++d) { float v = sx[tid + d]; s1 += v; s2 += v * v; }
  oS [row * HW + j0 + tid] = s1;
  oS2[row * HW + j0 + tid] = s2;
}

// Horizontal 59-wide box sum of two independent planes (A, b).
__global__ __launch_bounds__(256) void k_gf_hpass2(
    const float* __restrict__ inA, const float* __restrict__ inB,
    float* __restrict__ oA, float* __restrict__ oB) {
  __shared__ float sa[256 + WIN - 1];
  __shared__ float sb[256 + WIN - 1];
  const int row = blockIdx.y, j0 = blockIdx.x * 256, tid = threadIdx.x;
  for (int e = tid; e < 256 + WIN - 1; e += 256) {
    int gj = j0 - R + e;
    bool in = (gj >= 0 && gj < HW);
    sa[e] = in ? inA[row * HW + gj] : 0.0f;
    sb[e] = in ? inB[row * HW + gj] : 0.0f;
  }
  __syncthreads();
  float s1 = 0.0f, s2 = 0.0f;
  for (int d = 0; d < WIN; ++d) { s1 += sa[tid + d]; s2 += sb[tid + d]; }
  oA[row * HW + j0 + tid] = s1;
  oB[row * HW + j0 + tid] = s2;
}

// Vertical box of (Sx, Sxx) -> A, b  (self-guided: cov == var).
__global__ __launch_bounds__(256) void k_gf_vpass_ab(
    const float* __restrict__ Sx, const float* __restrict__ Sxx,
    float* __restrict__ oA, float* __restrict__ oB) {
  const int j = blockIdx.x * 256 + threadIdx.x;
  const int i = blockIdx.y;
  const int lo = (i - R < 0) ? 0 : i - R;
  const int hi = (i + R > HW - 1) ? HW - 1 : i + R;
  float s1 = 0.0f, s2 = 0.0f;
  for (int r = lo; r <= hi; ++r) {
    __builtin_prefetch(&Sx[(r + 8) * HW + j], 0, 1);   // global_prefetch_b8
    s1 += Sx [r * HW + j];
    s2 += Sxx[r * HW + j];
  }
  const int jlo = (j - R < 0) ? 0 : j - R;
  const int jhi = (j + R > HW - 1) ? HW - 1 : j + R;
  const float Ninv = 1.0f / ((float)(jhi - jlo + 1) * (float)(hi - lo + 1));
  const float mean = s1 * Ninv;
  const float var  = s2 * Ninv - mean * mean;
  const float A    = var / fmaxf(var, 1e-9f);
  oA[i * HW + j] = A;
  oB[i * HW + j] = mean - A * mean;
}

// Vertical box of (box(A), box(b)) -> q; accumulate sum|smoothedTarget - q|.
__global__ __launch_bounds__(256) void k_gf_final(
    const float* __restrict__ SA, const float* __restrict__ Sb,
    const float* __restrict__ x, const float* __restrict__ st,
    double* __restrict__ acc) {
  __shared__ float red[256];
  const int j = blockIdx.x * 256 + threadIdx.x;
  const int i = blockIdx.y;
  const int lo = (i - R < 0) ? 0 : i - R;
  const int hi = (i + R > HW - 1) ? HW - 1 : i + R;
  float s1 = 0.0f, s2 = 0.0f;
  for (int r = lo; r <= hi; ++r) {
    s1 += SA[r * HW + j];
    s2 += Sb[r * HW + j];
  }
  const int jlo = (j - R < 0) ? 0 : j - R;
  const int jhi = (j + R > HW - 1) ? HW - 1 : j + R;
  const float Ninv = 1.0f / ((float)(jhi - jlo + 1) * (float)(hi - lo + 1));
  const float q = (s1 * Ninv) * x[i * HW + j] + s2 * Ninv;
  reduce_add_256(fabsf(st[i * HW + j] - q), 1.0, acc, red);
}

// ------------------------------- K5: sampled -------------------------------
__global__ __launch_bounds__(256) void k_sampled(
    const float* __restrict__ tgt, const float* __restrict__ gnn,
    const float* __restrict__ mp, double* __restrict__ acc) {
  __shared__ float red[256];
  const int idx = blockIdx.x * 256 + threadIdx.x;   // 384*384 = 147456 exact
  const int si = idx / 384, sj = idx % 384;
  const int off = (si * 8) * HW + sj * 8;
  reduce_add_256(fabsf(tgt[off] - gnn[off]) * mp[off], 1.0, acc, red);
}

// --------------------------------- K6: TV ----------------------------------
__global__ __launch_bounds__(256) void k_tv(
    const float* __restrict__ gnn, const float* __restrict__ lr,
    const float* __restrict__ tgt, double* __restrict__ acc) {
  __shared__ float red[256];
  const float* x = (blockIdx.z == 0) ? gnn : lr;
  const int idx = blockIdx.x * 256 + threadIdx.x;   // exactly NPIX
  const int i = idx / HW, j = idx % HW;
  const float xv = x[idx];
  float local = 0.0f;
  if (j + 1 < HW) local += fabsf(xv - x[idx + 1]);                 // conv(x, Dx)
  if (i + 1 < HW) {                                                // conv(x-t, Dy)
    local += fabsf((xv - tgt[idx]) - (x[idx + HW] - tgt[idx + HW]));
  }
  reduce_add_256(local, 0.1, acc, red);   // LAMBDA_TV
}

// ----------------------- K7: Hessian via bf16 WMMA -------------------------
// D = A x B + C, M=16 output pixels of a row strip, N=16 channels
// (ch0: Gxx*x, ch1: Gyy*(x-t), ch2: Gxy*(x-t) [x2 weight], rest zero).
// K is PERMUTED so addressing needs no div/mod: each source's 13x13 kernel is
// padded to 13 rows x 16 cols (zero weight in cols 13..15); K = row*16 + col,
// sources concatenated: K in [0,208) = x-taps, [208,416) = (x-t)-taps.
// With the 13-block K loop fully unrolled, every A element is an LDS load at
// a compile-time immediate offset from one per-lane base. Inputs are split
// hi/lo bf16 (x = hi + lo) for ~fp32 accuracy: 26 WMMAs per wave (208 per 8x16 tile).
__device__ __forceinline__ float bweight(int K, int n,
                                         const float* Gxx, const float* Gyy,
                                         const float* Gxy) {
  const int src = (K >= KPERSRC);
  const int rk  = K - (src ? KPERSRC : 0);
  const int row = rk >> 4, col = rk & 15;
  if (col >= KS) return 0.0f;
  const int t = row * KS + col;
  if (!src) return (n == 0) ? Gxx[t] : 0.0f;
  if (n == 1) return Gyy[t];
  if (n == 2) return Gxy[t];
  return 0.0f;
}

__global__ __launch_bounds__(256) void k_hessian_wmma(
    const float* __restrict__ gnn, const float* __restrict__ lr,
    const float* __restrict__ tgt,
    const float* __restrict__ Gxx, const float* __restrict__ Gyy,
    const float* __restrict__ Gxy, double* __restrict__ acc) {
  __shared__ unsigned short tUhi[TILE_ELEMS], tUlo[TILE_ELEMS];
  __shared__ unsigned short tVhi[TILE_ELEMS], tVlo[TILE_ELEMS];
  __shared__ unsigned int   bStage[KBLK * 32 * 8];  // per-lane packed B dwords
  __shared__ float          red[256];

  const int tid = threadIdx.x;
  const float* x = (blockIdx.z == 0) ? gnn : lr;
  const int i0 = blockIdx.y * 8;

  // Stage B once per block: lane l -> n=l&15, h=l>>4; VGPR p holds
  // B[K,n],B[K+1,n] with K = blk*32 + h*8 + 2p (+8 for p>=4), matching A.
  for (int idx = tid; idx < KBLK * 32 * 8; idx += 256) {
    int blk = idx >> 8, rem = idx & 255;
    int lane = rem >> 3, p = rem & 7;
    int h = lane >> 4, n = lane & 15;
    int k = blk * 32 + h * 8 + 2 * p + (p >= 4 ? 8 : 0);
    unsigned lo16 = f32_to_bf16_bits(bweight(k,     n, Gxx, Gyy, Gxy));
    unsigned hi16 = f32_to_bf16_bits(bweight(k + 1, n, Gxx, Gyy, Gxy));
    bStage[idx] = (hi16 << 16) | lo16;
  }

  const int wave = tid >> 5;        // output row within tile
  const int lane = tid & 31;
  const int h = lane >> 4;
  const int m = lane & 15;          // A-matrix row = output pixel index
  const int baseE = wave * TILE_COLS + m + h * 8;   // per-lane A base element
  const int ch = lane & 15;
  const float cw = (ch == 2) ? 2.0f : ((ch < 2) ? 1.0f : 0.0f);  // iso Gxy x2
  const int irow = i0 + wave;

  float local = 0.0f;
  for (int jt = 0; jt < NTJ; ++jt) {
    const int j0 = (blockIdx.x * NTJ + jt) * 16;
    __syncthreads();   // tiles from previous sweep no longer in use
    // Stage input tiles (clamped loads feed only masked-out outputs).
    for (int e = tid; e < TILE_ELEMS; e += 256) {
      int a = e / TILE_COLS, c = e % TILE_COLS;
      int gi = i0 + a; if (gi > HW - 1) gi = HW - 1;
      int gj = j0 + c; if (gj > HW - 1) gj = HW - 1;
      float xv = x[gi * HW + gj];
      float vv = xv - tgt[gi * HW + gj];
      unsigned short uh = f32_to_bf16_bits(xv);
      unsigned short vh = f32_to_bf16_bits(vv);
      tUhi[e] = uh; tUlo[e] = f32_to_bf16_bits(xv - bf16_bits_to_f32(uh));
      tVhi[e] = vh; tVlo[e] = f32_to_bf16_bits(vv - bf16_bits_to_f32(vh));
    }
    __syncthreads();

    v8f d = {0.f, 0.f, 0.f, 0.f, 0.f, 0.f, 0.f, 0.f};
    #pragma unroll
    for (int blk = 0; blk < KBLK; ++blk) {
      // Two 16-wide K runs per block; each run has a fixed kernel row/source.
      const int K0 = blk * 32, K1 = K0 + 16;
      const int src0 = (K0 >= KPERSRC), src1 = (K1 >= KPERSRC);
      const int row0 = (K0 - (src0 ? KPERSRC : 0)) >> 4;
      const int row1 = (K1 - (src1 ? KPERSRC : 0)) >> 4;
      const unsigned short* p0hi = src0 ? tVhi : tUhi;
      const unsigned short* p0lo = src0 ? tVlo : tUlo;
      const unsigned short* p1hi = src1 ? tVhi : tUhi;
      const unsigned short* p1lo = src1 ? tVlo : tUlo;
      const int e0 = baseE + row0 * TILE_COLS;   // + const -> imm offsets
      const int e1 = baseE + row1 * TILE_COLS;

      v16bf Ahi, Alo, B;
      #pragma unroll
      for (int p = 0; p < 4; ++p) {              // K_local = h*8 + 2p (+1)
        Ahi[2 * p]     = bits_to_bf16(p0hi[e0 + 2 * p]);
        Ahi[2 * p + 1] = bits_to_bf16(p0hi[e0 + 2 * p + 1]);
        Alo[2 * p]     = bits_to_bf16(p0lo[e0 + 2 * p]);
        Alo[2 * p + 1] = bits_to_bf16(p0lo[e0 + 2 * p + 1]);
      }
      #pragma unroll
      for (int p = 4; p < 8; ++p) {              // K_local = 16 + h*8 + 2(p-4)
        Ahi[2 * p]     = bits_to_bf16(p1hi[e1 + 2 * (p - 4)]);
        Ahi[2 * p + 1] = bits_to_bf16(p1hi[e1 + 2 * (p - 4) + 1]);
        Alo[2 * p]     = bits_to_bf16(p1lo[e1 + 2 * (p - 4)]);
        Alo[2 * p + 1] = bits_to_bf16(p1lo[e1 + 2 * (p - 4) + 1]);
      }
      #pragma unroll
      for (int p = 0; p < 8; ++p) {
        unsigned w = bStage[(blk * 32 + lane) * 8 + p];
        B[2 * p]     = bits_to_bf16((unsigned short)(w & 0xFFFFu));
        B[2 * p + 1] = bits_to_bf16((unsigned short)(w >> 16));
      }
      d = __builtin_amdgcn_wmma_f32_16x16x32_bf16(false, Ahi, false, B,
                                                  (short)0, d, false, false);
      d = __builtin_amdgcn_wmma_f32_16x16x32_bf16(false, Alo, false, B,
                                                  (short)0, d, false, false);
    }

    // D layout: lane -> channel n = lane&15; VGPR p -> pixel (lane>>4)*8 + p.
    if (irow < OW) {
      #pragma unroll
      for (int p = 0; p < 8; ++p) {
        int j = j0 + h * 8 + p;
        if (j < OW) local += cw * fabsf(d[p]);
      }
    }
  }
  __syncthreads();
  reduce_add_256(local, 0.5, acc, red);   // LAMBDA_HESSIAN
}

// ------------------------------ launcher -----------------------------------
extern "C" void kernel_launch(void* const* d_in, const int* in_sizes, int n_in,
                              void* d_out, int out_size, void* d_ws, size_t ws_size,
                              hipStream_t stream) {
  (void)in_sizes; (void)n_in; (void)out_size; (void)ws_size;
  const float* raw   = (const float*)d_in[0];   // outputGNNraw
  const float* gnn   = (const float*)d_in[1];   // outputGNN
  const float* lr    = (const float*)d_in[2];   // outputLR
  const float* st    = (const float*)d_in[3];   // smoothedTarget
  const float* tgt   = (const float*)d_in[4];   // targets
  const float* mp    = (const float*)d_in[5];   // map
  const float* Gxx   = (const float*)d_in[8];   // DGaussxx (13x13)
  const float* Gyy   = (const float*)d_in[9];   // DGaussyy
  const float* Gxy   = (const float*)d_in[10];  // DGaussxy

  double* acc = (double*)d_ws;
  float*  B0  = (float*)((char*)d_ws + 256);
  float*  B1  = B0 + NPIX;
  float*  B2  = B1 + NPIX;
  float*  B3  = B2 + NPIX;

  const dim3 blk(256);
  const dim3 gRows(HW / 256, HW);          // 12 x 3072

  k_init_acc<<<1, 64, 0, stream>>>(acc);

  // Guided filter + sum|st - q|
  k_gf_hpass_sq<<<gRows, blk, 0, stream>>>(raw, B0, B1);
  k_gf_vpass_ab<<<gRows, blk, 0, stream>>>(B0, B1, B2, B3);
  k_gf_hpass2 <<<gRows, blk, 0, stream>>>(B2, B3, B0, B1);
  k_gf_final  <<<gRows, blk, 0, stream>>>(B0, B1, raw, st, acc);

  // Sampled masked L1
  k_sampled<<<(384 * 384) / 256, blk, 0, stream>>>(tgt, gnn, mp, acc);

  // TV (both images)
  k_tv<<<dim3(NPIX / 256, 1, 2), blk, 0, stream>>>(gnn, lr, tgt, acc);

  // Hessian convs via WMMA (both images); each block sweeps NTJ column tiles.
  k_hessian_wmma<<<dim3((OW + 16 * NTJ - 1) / (16 * NTJ), (OW + 7) / 8, 2),
                   blk, 0, stream>>>(gnn, lr, tgt, Gxx, Gyy, Gxy, acc);

  k_finalize<<<1, 64, 0, stream>>>(acc, (float*)d_out);
}
